// InpaintGenerator_40200893890678
// MI455X (gfx1250) — compile-verified, matrix operand order checked by reading, not run on hardware
//
#include <hip/hip_runtime.h>
#include <hip/hip_bf16.h>
#include <math.h>

typedef _Float16 f16;
typedef __attribute__((ext_vector_type(8)))  _Float16 v8h;
typedef __attribute__((ext_vector_type(16))) _Float16 v16h;
typedef __attribute__((ext_vector_type(8)))  float    v8f;

#define ACT_NONE  0
#define ACT_RELU  1
#define ACT_LRELU 2
#define ACT_TANH  3

static inline unsigned cdiv(size_t a, size_t b) { return (unsigned)((a + b - 1) / b); }
static inline int pad32(int k) { return (k + 31) & ~31; }

// ---------------------------------------------------------------------------
// Kernels
// ---------------------------------------------------------------------------

// f32 [M,K] -> f16 [M,Kp] zero-padded (weight conversion)
__global__ void k_cast_f16_pad(const float* __restrict__ s, f16* __restrict__ d,
                               int M, int K, int Kp) {
    size_t i = (size_t)blockIdx.x * blockDim.x + threadIdx.x;
    if (i >= (size_t)M * Kp) return;
    int c = (int)(i % Kp);
    int r = (int)(i / Kp);
    d[i] = (c < K) ? (f16)s[(size_t)r * K + c] : (f16)0.f;
}

// im2col producing B^T [L, Dp] in f16 (zero-padded cols D..Dp),
// optional per-row (per-patch) norm divide. d = c*k*k + ki*k + kj ; p = oy*OW+ox
__global__ void k_im2col_f16(const float* __restrict__ x, f16* __restrict__ dst,
                             const float* __restrict__ rnorm,
                             int C, int H, int W, int k, int stride, int pad, int dil,
                             int OH, int OW, int Dp) {
    size_t idx = (size_t)blockIdx.x * blockDim.x + threadIdx.x;
    int D = C * k * k;
    size_t total = (size_t)OH * OW * Dp;
    if (idx >= total) return;
    int d = (int)(idx % Dp);
    if (d >= D) { dst[idx] = (f16)0.f; return; }
    size_t p = idx / Dp;
    int c = d / (k * k); int rem = d % (k * k); int ki = rem / k; int kj = rem % k;
    int oy = (int)(p / OW), ox = (int)(p % OW);
    int y = oy * stride - pad + ki * dil;
    int xx = ox * stride - pad + kj * dil;
    float v = 0.f;
    if (y >= 0 && y < H && xx >= 0 && xx < W) v = x[((size_t)c * H + y) * W + xx];
    if (rnorm) v /= rnorm[p];
    dst[idx] = (f16)v;
}

// patches transposed: dst[(c*k+i)*k+j, p] f16, row-major ld = Pp (cols P..Pp zeroed)
__global__ void k_patchesT_f16(const float* __restrict__ x, f16* __restrict__ dst,
                               int C, int H, int W, int k, int stride, int pad,
                               int OH, int OW, int Pp) {
    size_t idx = (size_t)blockIdx.x * blockDim.x + threadIdx.x;
    int P = OH * OW;
    size_t total = (size_t)C * k * k * Pp;
    if (idx >= total) return;
    int p = (int)(idx % Pp);
    if (p >= P) { dst[idx] = (f16)0.f; return; }
    int r = (int)(idx / Pp);
    int c = r / (k * k); int rem = r % (k * k); int i = rem / k; int j = rem % k;
    int oy = p / OW, ox = p % OW;
    int y = oy * stride - pad + i, xx = ox * stride - pad + j;
    float v = 0.f;
    if (y >= 0 && y < H && xx >= 0 && xx < W) v = x[((size_t)c * H + y) * W + xx];
    dst[idx] = (f16)v;
}

// WMMA GEMM: C[M,N] f32 = A[M,K] f16 * B, with B supplied transposed as Bt[N,K].
// K must be a multiple of 32 (buffers zero-padded). One wave -> 16x64 tile:
// one A fragment reused by four B fragments, 4 v_wmma per K-step, branch-free
// inner loop with 128-bit loads (rows out of range are clamped; their garbage
// lands only in accumulator slots that are never stored).
__global__ void k_gemm_wmma(const f16* __restrict__ A, const f16* __restrict__ Bt,
                            float* __restrict__ C, int M, int N, int K) {
    const int lane = threadIdx.x & 31;
    const int nTile = (int)blockIdx.x * 64;
    const int mTile = (int)blockIdx.y * 16;
    const int hl  = lane >> 4;
    const int l15 = lane & 15;
    int rA = mTile + l15; if (rA >= M) rA = M - 1;
    const f16* pA = A + (size_t)rA * K + hl * 8;
    const f16* pB0; const f16* pB1; const f16* pB2; const f16* pB3;
    {
        int r0 = nTile      + l15; if (r0 >= N) r0 = N - 1;
        int r1 = nTile + 16 + l15; if (r1 >= N) r1 = N - 1;
        int r2 = nTile + 32 + l15; if (r2 >= N) r2 = N - 1;
        int r3 = nTile + 48 + l15; if (r3 >= N) r3 = N - 1;
        pB0 = Bt + (size_t)r0 * K + hl * 8;
        pB1 = Bt + (size_t)r1 * K + hl * 8;
        pB2 = Bt + (size_t)r2 * K + hl * 8;
        pB3 = Bt + (size_t)r3 * K + hl * 8;
    }
    v8f acc0 = {}, acc1 = {}, acc2 = {}, acc3 = {};
    for (int kt = 0; kt < K; kt += 32) {
        v8h a0 = *(const v8h*)pA;
        v8h a1 = *(const v8h*)(pA + 16);
        v16h a = __builtin_shufflevector(a0, a1, 0,1,2,3,4,5,6,7,8,9,10,11,12,13,14,15);
        __builtin_prefetch((const void*)(pA + 512), 0, 1);
        pA += 32;

        v8h b0l = *(const v8h*)pB0, b0h = *(const v8h*)(pB0 + 16);
        v8h b1l = *(const v8h*)pB1, b1h = *(const v8h*)(pB1 + 16);
        v8h b2l = *(const v8h*)pB2, b2h = *(const v8h*)(pB2 + 16);
        v8h b3l = *(const v8h*)pB3, b3h = *(const v8h*)(pB3 + 16);
        __builtin_prefetch((const void*)(pB0 + 512), 0, 1);
        __builtin_prefetch((const void*)(pB3 + 512), 0, 1);
        pB0 += 32; pB1 += 32; pB2 += 32; pB3 += 32;

        v16h b;
        b = __builtin_shufflevector(b0l, b0h, 0,1,2,3,4,5,6,7,8,9,10,11,12,13,14,15);
        acc0 = __builtin_amdgcn_wmma_f32_16x16x32_f16(false, a, false, b, (short)0, acc0, false, false);
        b = __builtin_shufflevector(b1l, b1h, 0,1,2,3,4,5,6,7,8,9,10,11,12,13,14,15);
        acc1 = __builtin_amdgcn_wmma_f32_16x16x32_f16(false, a, false, b, (short)0, acc1, false, false);
        b = __builtin_shufflevector(b2l, b2h, 0,1,2,3,4,5,6,7,8,9,10,11,12,13,14,15);
        acc2 = __builtin_amdgcn_wmma_f32_16x16x32_f16(false, a, false, b, (short)0, acc2, false, false);
        b = __builtin_shufflevector(b3l, b3h, 0,1,2,3,4,5,6,7,8,9,10,11,12,13,14,15);
        acc3 = __builtin_amdgcn_wmma_f32_16x16x32_f16(false, a, false, b, (short)0, acc3, false, false);
    }
#pragma unroll
    for (int i = 0; i < 8; ++i) {
        int r = mTile + hl * 8 + i;
        if (r >= M) continue;
        float* Cr = C + (size_t)r * N;
        int c0 = nTile + l15;
        if (c0      < N) Cr[c0]      = acc0[i];
        if (c0 + 16 < N) Cr[c0 + 16] = acc1[i];
        if (c0 + 32 < N) Cr[c0 + 32] = acc2[i];
        if (c0 + 48 < N) Cr[c0 + 48] = acc3[i];
    }
}

// y[c, l] = act(y[c, l] + bias[c])
__global__ void k_bias_act(float* __restrict__ y, const float* __restrict__ b,
                           int C, int L, int act) {
    size_t idx = (size_t)blockIdx.x * blockDim.x + threadIdx.x;
    if (idx >= (size_t)C * L) return;
    int c = (int)(idx / L);
    float v = y[idx] + b[c];
    if (act == ACT_RELU)       v = fmaxf(v, 0.f);
    else if (act == ACT_LRELU) v = v > 0.f ? v : 0.2f * v;
    else if (act == ACT_TANH)  v = tanhf(v);
    y[idx] = v;
}

// bilinear resize, align_corners=True
__global__ void k_resize(const float* __restrict__ src, float* __restrict__ dst,
                         int C, int H, int W, int OH, int OW) {
    size_t idx = (size_t)blockIdx.x * blockDim.x + threadIdx.x;
    if (idx >= (size_t)C * OH * OW) return;
    int ox = (int)(idx % OW);
    int oy = (int)((idx / OW) % OH);
    int c  = (int)(idx / ((size_t)OW * OH));
    float ys = (OH > 1) ? oy * (float)(H - 1) / (OH - 1) : 0.f;
    float xs = (OW > 1) ? ox * (float)(W - 1) / (OW - 1) : 0.f;
    int ymax = H - 2 > 0 ? H - 2 : 0;
    int xmax = W - 2 > 0 ? W - 2 : 0;
    int y0 = (int)floorf(ys); y0 = y0 < 0 ? 0 : (y0 > ymax ? ymax : y0);
    int x0 = (int)floorf(xs); x0 = x0 < 0 ? 0 : (x0 > xmax ? xmax : x0);
    int y1 = y0 + 1 < H - 1 ? y0 + 1 : H - 1;
    int x1 = x0 + 1 < W - 1 ? x0 + 1 : W - 1;
    float wy = ys - y0, wx = xs - x0;
    const float* s = src + (size_t)c * H * W;
    float v = (1.f - wy) * ((1.f - wx) * s[y0 * W + x0] + wx * s[y0 * W + x1])
            +        wy  * ((1.f - wx) * s[y1 * W + x0] + wx * s[y1 * W + x1]);
    dst[idx] = v;
}

// mm[p] = (sum over kxk zero-padded mask patch == 0) ? 1 : 0
__global__ void k_valid_mask(const float* __restrict__ m, float* __restrict__ mm,
                             int H, int W, int k, int pad) {
    int p = blockIdx.x * blockDim.x + threadIdx.x;
    if (p >= H * W) return;
    int oy = p / W, ox = p % W;
    float s = 0.f;
    for (int i = 0; i < k; ++i) {
        int y = oy - pad + i; if (y < 0 || y >= H) continue;
        for (int j = 0; j < k; ++j) {
            int x = ox - pad + j; if (x < 0 || x >= W) continue;
            s += m[y * W + x];
        }
    }
    mm[p] = (s == 0.f) ? 1.f : 0.f;
}

// norms[p] = max(sqrt(sum over stride-1 patch x^2), 1e-4)
__global__ void k_patch_sqnorm(const float* __restrict__ x, float* __restrict__ norms,
                               int C, int H, int W, int k, int pad) {
    int p = blockIdx.x * blockDim.x + threadIdx.x;
    if (p >= H * W) return;
    int oy = p / W, ox = p % W;
    float s = 0.f;
    for (int c = 0; c < C; ++c)
        for (int i = 0; i < k; ++i) {
            int y = oy - pad + i; if (y < 0 || y >= H) continue;
            for (int j = 0; j < k; ++j) {
                int xx = ox - pad + j; if (xx < 0 || xx >= W) continue;
                float v = x[((size_t)c * H + y) * W + xx];
                s += v * v;
            }
        }
    norms[p] = fmaxf(sqrtf(s), 1e-4f);
}

// Row-wise over scoreT[p,0..L): t=score*mm*scale; softmax; out=max(soft*mm,1e-8)
// Written as f16 with row stride Lp (cols L..Lp zeroed).
__global__ void k_attn_softmax(const float* __restrict__ scoreT, const float* __restrict__ mm,
                               f16* __restrict__ attT, int L, int Lp, float scale) {
    __shared__ float red[256];
    int p = blockIdx.x, t = threadIdx.x;
    const float* row = scoreT + (size_t)p * L;
    float mx = -1e30f;
    for (int f = t; f < L; f += 256) mx = fmaxf(mx, row[f] * mm[f] * scale);
    red[t] = mx; __syncthreads();
    for (int s = 128; s > 0; s >>= 1) { if (t < s) red[t] = fmaxf(red[t], red[t + s]); __syncthreads(); }
    mx = red[0]; __syncthreads();
    float sum = 0.f;
    for (int f = t; f < L; f += 256) sum += __expf(row[f] * mm[f] * scale - mx);
    red[t] = sum; __syncthreads();
    for (int s = 128; s > 0; s >>= 1) { if (t < s) red[t] += red[t + s]; __syncthreads(); }
    float inv = 1.f / red[0];
    for (int f = t; f < Lp; f += 256) {
        float v = 0.f;
        if (f < L) v = fmaxf(__expf(row[f] * mm[f] * scale - mx) * inv * mm[f], 1e-8f);
        attT[(size_t)p * Lp + f] = (f16)v;
    }
}

// Transposed-conv overlap-add from U[(c*4+i)*4+j, py*w2+px] (kernel=4, rate=2, pad=1)
__global__ void k_overlap_add(const float* __restrict__ U, float* __restrict__ out,
                              int C, int h2, int w2) {
    int h1 = 2 * h2, w1 = 2 * w2;
    size_t idx = (size_t)blockIdx.x * blockDim.x + threadIdx.x;
    if (idx >= (size_t)C * h1 * w1) return;
    int x = (int)(idx % w1);
    int y = (int)((idx / w1) % h1);
    int c = (int)(idx / ((size_t)w1 * h1));
    int P = h2 * w2;
    float s = 0.f;
    for (int i = 0; i < 4; ++i) {
        int ty = y + 1 - i; if (ty < 0 || (ty & 1)) continue;
        int py = ty >> 1;   if (py >= h2) continue;
        for (int j = 0; j < 4; ++j) {
            int tx = x + 1 - j; if (tx < 0 || (tx & 1)) continue;
            int px = tx >> 1;   if (px >= w2) continue;
            s += U[((size_t)((c * 4 + i) * 4 + j)) * P + py * w2 + px];
        }
    }
    out[idx] = 0.25f * s;
}

// ---------------------------------------------------------------------------
// Host-side orchestration helpers
// ---------------------------------------------------------------------------

struct Alloc {
    char* base; size_t off; size_t cap;
    void* get(size_t bytes) {
        size_t o = (off + 255) & ~(size_t)255;
        off = o + bytes;
        return (void*)(base + o);
    }
};

static void gemm(hipStream_t st, const f16* A, const f16* Bt, float* C, int M, int N, int K) {
    dim3 grid(cdiv(N, 64), cdiv(M, 16));
    k_gemm_wmma<<<grid, 32, 0, st>>>(A, Bt, C, M, N, K);
}

// conv2d via im2col + WMMA GEMM + bias/activation. Wt:[Cout, pad32(Cin*k*k)] f16
static void conv2d(hipStream_t st, Alloc a, const float* x, int Cin, int H, int W,
                   const f16* Wt, const float* bias, int Cout,
                   int k, int stride, int pad, int dil, float* y, int act) {
    int OH = (H + 2 * pad - dil * (k - 1) - 1) / stride + 1;
    int OW = (W + 2 * pad - dil * (k - 1) - 1) / stride + 1;
    int L = OH * OW, Dp = pad32(Cin * k * k);
    f16* bt = (f16*)a.get((size_t)L * Dp * sizeof(f16));
    size_t tot = (size_t)L * Dp;
    k_im2col_f16<<<cdiv(tot, 256), 256, 0, st>>>(x, bt, nullptr, Cin, H, W,
                                                 k, stride, pad, dil, OH, OW, Dp);
    gemm(st, Wt, bt, y, Cout, L, Dp);
    k_bias_act<<<cdiv((size_t)Cout * L, 256), 256, 0, st>>>(y, bias, Cout, L, act);
}

// Contextual attention, single image. rate=2, kernel=4, scale=10.
static void atn_single(hipStream_t st, Alloc a,
                       const float* x1, int C1, int h1, int w1,
                       const float* x2, int C2, int h2, int w2,
                       const float* m2, int ksize, float* out) {
    int P = h2 * w2, Pp = pad32(P);
    int padw = (ksize == 1) ? 0 : 1;
    int Dp = pad32(C2 * ksize * ksize);
    float* mm    = (float*)a.get((size_t)P * 4);
    float* norms = (float*)a.get((size_t)P * 4);
    f16* wf      = (f16*)a.get((size_t)P * Dp * 2);
    f16* wn      = (f16*)a.get((size_t)P * Dp * 2);
    float* scoreT= (float*)a.get((size_t)P * P * 4);
    f16* attT    = (f16*)a.get((size_t)P * Pp * 2);
    f16* rawWt   = (f16*)a.get((size_t)C1 * 16 * Pp * 2);
    float* U     = (float*)a.get((size_t)C1 * 16 * P * 4);

    k_valid_mask<<<cdiv(P, 256), 256, 0, st>>>(m2, mm, h2, w2, ksize, padw);
    k_patch_sqnorm<<<cdiv(P, 128), 128, 0, st>>>(x2, norms, C2, h2, w2, ksize, padw);
    size_t tw = (size_t)P * Dp;
    k_im2col_f16<<<cdiv(tw, 256), 256, 0, st>>>(x2, wf, nullptr, C2, h2, w2,
                                                ksize, 1, padw, 1, h2, w2, Dp);
    k_im2col_f16<<<cdiv(tw, 256), 256, 0, st>>>(x2, wn, norms, C2, h2, w2,
                                                ksize, 1, padw, 1, h2, w2, Dp);
    gemm(st, wf, wn, scoreT, P, P, Dp);                 // scoreT[p,f] = wf[p].wn[f]
    k_attn_softmax<<<P, 256, 0, st>>>(scoreT, mm, attT, P, Pp, 10.0f);
    size_t tr = (size_t)C1 * 16 * Pp;
    k_patchesT_f16<<<cdiv(tr, 256), 256, 0, st>>>(x1, rawWt, C1, h1, w1, 4, 2, 1, h2, w2, Pp);
    gemm(st, rawWt, attT, U, C1 * 16, P, Pp);           // U[ci,p] = sum_f rawWt[ci,f]*attT[p,f]
    size_t to = (size_t)C1 * h1 * w1;
    k_overlap_add<<<cdiv(to, 256), 256, 0, st>>>(U, out, C1, h2, w2);
}

// ---------------------------------------------------------------------------
// Entry point
// ---------------------------------------------------------------------------

extern "C" void kernel_launch(void* const* d_in, const int* in_sizes, int n_in,
                              void* d_out, int out_size, void* d_ws, size_t ws_size,
                              hipStream_t stream) {
    (void)in_sizes; (void)n_in; (void)out_size;
    const int B = 2;
    Alloc a{(char*)d_ws, 0, ws_size};

    const float* img  = (const float*)d_in[0];   // [2,3,256,256]
    const float* mask = (const float*)d_in[1];   // [2,1,256,256]

    // ---- convert conv weights to f16 [M, pad32(K)] (persistent in ws) ----
    auto cvt = [&](int idx, int M, int K) -> const f16* {
        int Kp = pad32(K);
        f16* d = (f16*)a.get((size_t)M * Kp * 2);
        k_cast_f16_pad<<<cdiv((size_t)M * Kp, 256), 256, 0, stream>>>(
            (const float*)d_in[idx], d, M, K, Kp);
        return d;
    };
    auto bias = [&](int idx) -> const float* { return (const float*)d_in[idx]; };

    const f16 *dw01w=cvt(2,32,36),     *dw02w=cvt(4,64,288),    *dw03w=cvt(6,128,576);
    const f16 *dw04w=cvt(8,256,1152),  *dw05w=cvt(10,512,2304), *dw06w=cvt(12,512,4608);
    const f16 *up05w=cvt(14,512,4608), *up04w=cvt(16,256,9216), *up03w=cvt(18,128,4608);
    const f16 *up02w=cvt(20,64,2304),  *up01w=cvt(22,32,1152);
    const f16 *tg5w=cvt(24,3,1024), *tg4w=cvt(26,3,512), *tg3w=cvt(28,3,256);
    const f16 *tg2w=cvt(30,3,128),  *tg1w=cvt(32,3,64);
    const f16 *dec0w=cvt(34,32,576), *dec1w=cvt(36,3,288);
    const f16* at04w[4]; const float* at04b[4];
    const f16* at03w[4]; const float* at03b[4];
    const f16* at02w[4]; const float* at02b[4];
    const f16* at01w[4]; const float* at01b[4];
    for (int g = 0; g < 4; ++g) {
        at04w[g] = cvt(38 + 2 * g, 64, 2304); at04b[g] = bias(39 + 2 * g);
        at03w[g] = cvt(46 + 2 * g, 32, 1152); at03b[g] = bias(47 + 2 * g);
        at02w[g] = cvt(54 + 2 * g, 16, 576);  at02b[g] = bias(55 + 2 * g);
        at01w[g] = cvt(62 + 2 * g, 8,  288);  at01b[g] = bias(63 + 2 * g);
    }

    // ---- persistent feature maps ----
    auto fa = [&](size_t n) -> float* { return (float*)a.get(n * 4); };
    float* x0  = fa((size_t)B * 4 * 256 * 256);
    float* x1  = fa((size_t)B * 32 * 128 * 128);
    float* x2  = fa((size_t)B * 64 * 64 * 64);
    float* x3  = fa((size_t)B * 128 * 32 * 32);
    float* x4  = fa((size_t)B * 256 * 16 * 16);
    float* x5  = fa((size_t)B * 512 * 8 * 8);
    float* x6  = fa((size_t)B * 512 * 4 * 4);
    float* x5a = fa((size_t)B * 512 * 8 * 8);
    float* x4a = fa((size_t)B * 256 * 16 * 16);
    float* x3a = fa((size_t)B * 128 * 32 * 32);
    float* x2a = fa((size_t)B * 64 * 64 * 64);
    float* x1a = fa((size_t)B * 32 * 128 * 128);
    float* upx5= fa((size_t)B * 512 * 8 * 8);
    float* upx4= fa((size_t)B * 256 * 16 * 16);
    float* upx3= fa((size_t)B * 128 * 32 * 32);
    float* upx2= fa((size_t)B * 64 * 64 * 64);
    float* upx1= fa((size_t)B * 32 * 128 * 128);
    float* c5  = fa((size_t)B * 1024 * 8 * 8);
    float* c4  = fa((size_t)B * 512 * 16 * 16);
    float* c3  = fa((size_t)B * 256 * 32 * 32);
    float* c2  = fa((size_t)B * 128 * 64 * 64);
    float* c1  = fa((size_t)B * 64 * 128 * 128);
    float* hbuf= fa((size_t)B * 32 * 256 * 256);

    Alloc scratch = a;   // transient region (by-value Alloc copies auto-rewind)

    float* out  = (float*)d_out;            // [2,3,256,256]
    float* img1 = out  + 393216;            // [2,3,128,128]
    float* img2 = img1 + 98304;             // [2,3,64,64]
    float* img3 = img2 + 24576;             // [2,3,32,32]
    float* img4 = img3 + 6144;              // [2,3,16,16]
    float* img5 = img4 + 1536;              // [2,3,8,8]

    auto conv_b = [&](const float* X, int Cin, int H, int W, const f16* Wt, const float* bb,
                      int Cout, int k, int s, int pad, int dil, float* Y, int act) {
        int OH = (H + 2 * pad - dil * (k - 1) - 1) / s + 1;
        int OW = (W + 2 * pad - dil * (k - 1) - 1) / s + 1;
        for (int b = 0; b < B; ++b)
            conv2d(stream, scratch, X + (size_t)b * Cin * H * W, Cin, H, W,
                   Wt, bb, Cout, k, s, pad, dil, Y + (size_t)b * Cout * OH * OW, act);
    };
    auto up_conv = [&](const float* X, int C, int H, int W, const f16* Wt, const float* bb,
                       int Cout, float* Y, int act) {
        for (int b = 0; b < B; ++b) {
            Alloc sb = scratch;
            float* u = (float*)sb.get((size_t)C * 4 * H * W * 4);
            size_t tot = (size_t)C * 2 * H * 2 * W;
            k_resize<<<cdiv(tot, 256), 256, 0, stream>>>(X + (size_t)b * C * H * W,
                                                         u, C, H, W, 2 * H, 2 * W);
            conv2d(stream, sb, u, C, 2 * H, 2 * W, Wt, bb, Cout, 3, 1, 1, 1,
                   Y + (size_t)b * Cout * 2 * H * 2 * W, act);
        }
    };
    auto cat2 = [&](const float* A1, int Ca, const float* A2, int Cb, int H, int W, float* D) {
        size_t na = (size_t)Ca * H * W, nb = (size_t)Cb * H * W;
        for (int b = 0; b < B; ++b) {
            hipMemcpyAsync(D + (size_t)b * (na + nb),      A1 + (size_t)b * na, na * 4,
                           hipMemcpyDeviceToDevice, stream);
            hipMemcpyAsync(D + (size_t)b * (na + nb) + na, A2 + (size_t)b * nb, nb * 4,
                           hipMemcpyDeviceToDevice, stream);
        }
    };
    auto atn = [&](const float* X1, int C1, int h1, int w1,
                   const float* X2, int C2, int h2, int w2, int ksize,
                   const f16* const* fw, const float* const* fb, float* OUT) {
        const int rates[4] = {1, 2, 4, 8};
        for (int b = 0; b < B; ++b) {
            Alloc ab = scratch;
            float* m2 = (float*)ab.get((size_t)h2 * w2 * 4);
            k_resize<<<cdiv((size_t)h2 * w2, 256), 256, 0, stream>>>(
                mask + (size_t)b * 256 * 256, m2, 1, 256, 256, h2, w2);
            float* outp = OUT + (size_t)b * C1 * h1 * w1;
            float* y = fw ? (float*)ab.get((size_t)C1 * h1 * w1 * 4) : outp;
            atn_single(stream, ab, X1 + (size_t)b * C1 * h1 * w1, C1, h1, w1,
                       X2 + (size_t)b * C2 * h2 * w2, C2, h2, w2, m2, ksize, y);
            if (fw) {
                for (int g = 0; g < 4; ++g)
                    conv2d(stream, ab, y, C1, h1, w1, fw[g], fb[g], C1 / 4,
                           3, 1, rates[g], rates[g],
                           outp + (size_t)g * (C1 / 4) * h1 * w1, ACT_RELU);
            }
        }
    };

    // ---- x = cat(img, mask) ----
    for (int b = 0; b < B; ++b) {
        hipMemcpyAsync(x0 + (size_t)b * 4 * 65536, img + (size_t)b * 3 * 65536,
                       (size_t)3 * 65536 * 4, hipMemcpyDeviceToDevice, stream);
        hipMemcpyAsync(x0 + (size_t)b * 4 * 65536 + 3 * 65536, mask + (size_t)b * 65536,
                       (size_t)65536 * 4, hipMemcpyDeviceToDevice, stream);
    }

    // ---- encoder ----
    conv_b(x0, 4,   256, 256, dw01w, bias(3),  32,  3, 2, 1, 1, x1, ACT_LRELU);
    conv_b(x1, 32,  128, 128, dw02w, bias(5),  64,  3, 2, 1, 1, x2, ACT_LRELU);
    conv_b(x2, 64,  64,  64,  dw03w, bias(7),  128, 3, 2, 1, 1, x3, ACT_LRELU);
    conv_b(x3, 128, 32,  32,  dw04w, bias(9),  256, 3, 2, 1, 1, x4, ACT_LRELU);
    conv_b(x4, 256, 16,  16,  dw05w, bias(11), 512, 3, 2, 1, 1, x5, ACT_LRELU);
    conv_b(x5, 512, 8,   8,   dw06w, bias(13), 512, 3, 2, 1, 1, x6, ACT_RELU);

    // ---- contextual attention pyramid ----
    atn(x5, 512, 8,   8,   x6,  512, 4,  4,  1, nullptr, nullptr, x5a);
    atn(x4, 256, 16,  16,  x5a, 512, 8,  8,  3, at04w, at04b, x4a);
    atn(x3, 128, 32,  32,  x4a, 256, 16, 16, 3, at03w, at03b, x3a);
    atn(x2, 64,  64,  64,  x3a, 128, 32, 32, 3, at02w, at02b, x2a);
    atn(x1, 32,  128, 128, x2a, 64,  64, 64, 3, at01w, at01b, x1a);

    // ---- decoder ----
    up_conv(x6, 512, 4, 4, up05w, bias(15), 512, upx5, ACT_RELU);
    cat2(upx5, 512, x5a, 512, 8, 8, c5);
    conv_b(c5, 1024, 8, 8, tg5w, bias(25), 3, 1, 1, 0, 1, img5, ACT_TANH);
    up_conv(c5, 1024, 8, 8, up04w, bias(17), 256, upx4, ACT_RELU);
    cat2(upx4, 256, x4a, 256, 16, 16, c4);
    conv_b(c4, 512, 16, 16, tg4w, bias(27), 3, 1, 1, 0, 1, img4, ACT_TANH);
    up_conv(c4, 512, 16, 16, up03w, bias(19), 128, upx3, ACT_RELU);
    cat2(upx3, 128, x3a, 128, 32, 32, c3);
    conv_b(c3, 256, 32, 32, tg3w, bias(29), 3, 1, 1, 0, 1, img3, ACT_TANH);
    up_conv(c3, 256, 32, 32, up02w, bias(21), 64, upx2, ACT_RELU);
    cat2(upx2, 64, x2a, 64, 64, 64, c2);
    conv_b(c2, 128, 64, 64, tg2w, bias(31), 3, 1, 1, 0, 1, img2, ACT_TANH);
    up_conv(c2, 128, 64, 64, up01w, bias(23), 32, upx1, ACT_RELU);
    cat2(upx1, 32, x1a, 32, 128, 128, c1);
    conv_b(c1, 64, 128, 128, tg1w, bias(33), 3, 1, 1, 0, 1, img1, ACT_TANH);

    up_conv(c1, 64, 128, 128, dec0w, bias(35), 32, hbuf, ACT_RELU);
    conv_b(hbuf, 32, 256, 256, dec1w, bias(37), 3, 3, 1, 1, 1, out, ACT_TANH);
}
